// CausalSelfAttention_13941463842848
// MI455X (gfx1250) — matrix operand order repaired and checked by reading, not compiled, not run on hardware
//
#include <hip/hip_runtime.h>

#define MODEL_DIM 2048
#define NUM_HEADS 16
#define NUM_KV_HEADS 4
#define HEAD_DIM 128
#define SEQ 4096
#define KV_DIM (NUM_KV_HEADS * HEAD_DIM)  // 512
#define WINDOW 1024

typedef __bf16 bf16_t;
typedef __bf16 v16bf __attribute__((ext_vector_type(16)));
typedef __bf16 v8bf  __attribute__((ext_vector_type(8)));
typedef float  v8f   __attribute__((ext_vector_type(8)));
typedef int    v4i_  __attribute__((vector_size(16)));

union V16 { v16bf v; v8bf h[2]; };

#if __has_builtin(__builtin_amdgcn_global_load_async_to_lds_b128) && \
    __has_builtin(__builtin_amdgcn_s_wait_asynccnt)
#define USE_ASYNC 1
#else
#define USE_ASYNC 0
#endif

// 16-byte global -> LDS copy; async (VGPR-bypassing, ASYNCcnt-tracked) when available.
__device__ __forceinline__ void cp16(bf16_t* lds, const bf16_t* g) {
#if USE_ASYNC
  __builtin_amdgcn_global_load_async_to_lds_b128(
      (__attribute__((address_space(1))) v4i_*)g,
      (__attribute__((address_space(3))) v4i_*)lds, 0, 0);
#else
  *(uint4*)lds = *(const uint4*)g;
#endif
}

__device__ __forceinline__ void wait_lds_fill() {
#if USE_ASYNC
  __builtin_amdgcn_s_wait_asynccnt(0);
#endif
  __syncthreads();
}

// ---------------------------------------------------------------- cast fp32 -> bf16
__global__ void cast_f32_bf16(const float* __restrict__ in, bf16_t* __restrict__ out, int n) {
  int i = blockIdx.x * blockDim.x + threadIdx.x;
  int stride = gridDim.x * blockDim.x;
  for (; i < n; i += stride) out[i] = (bf16_t)in[i];
}

// ---------------------------------------------------------------- WMMA GEMM
// C(M,N) = A(M,K) * W(N,K)^T ; A,W bf16 row-major; block tile 128x128, BK=32,
// double-buffered LDS with async global->LDS tile fills.
// 8 waves, wave tile 32x64 (2x4 frags of 16x16).
template <bool F32OUT>
__global__ __launch_bounds__(256, 1)
void gemm_bf16(const bf16_t* __restrict__ A, const bf16_t* __restrict__ W,
               void* __restrict__ Cout, int M, int N, int K) {
  __shared__ bf16_t As[2][128][40];  // +8 pad, row stride 80B (16B aligned)
  __shared__ bf16_t Bs[2][128][40];

  const int tid    = threadIdx.x;
  const int wave   = tid >> 5;
  const int lane   = tid & 31;
  const int laneLo = lane & 15;
  const int hi     = lane >> 4;
  const int m0 = blockIdx.y * 128;
  const int n0 = blockIdx.x * 128;
  const int wm = (wave >> 1) * 32;
  const int wn = (wave & 1) * 64;

  v8f acc[2][4];
  for (int i = 0; i < 2; i++)
    for (int j = 0; j < 4; j++)
      for (int r = 0; r < 8; r++) acc[i][j][r] = 0.0f;

  const int ldRow = tid >> 1;
  const int ldSeg = (tid & 1) * 16;
  const bf16_t* gA = A + (size_t)(m0 + ldRow) * K + ldSeg;
  const bf16_t* gW = W + (size_t)(n0 + ldRow) * K + ldSeg;

  auto load_tile = [&](int b, int k0) {
    cp16(&As[b][ldRow][ldSeg],     gA + k0);
    cp16(&As[b][ldRow][ldSeg + 8], gA + k0 + 8);
    cp16(&Bs[b][ldRow][ldSeg],     gW + k0);
    cp16(&Bs[b][ldRow][ldSeg + 8], gW + k0 + 8);
  };

  load_tile(0, 0);
  wait_lds_fill();

  int buf = 0;
  for (int k0 = 0; k0 < K; k0 += 32) {
    const bool more = (k0 + 32) < K;
    if (more) load_tile(buf ^ 1, k0 + 32);            // overlap with compute
    if (k0 + 128 < K) {                               // far prefetch into L2
      __builtin_prefetch(gA + k0 + 128, 0, 1);
      __builtin_prefetch(gW + k0 + 128, 0, 1);
    }

    V16 af[2], bfrag[4];
    for (int i = 0; i < 2; i++) {
      const int row = wm + i * 16 + laneLo;
      af[i].h[0] = *(const v8bf*)&As[buf][row][8 * hi];       // K = 0..7  (+8 if hi)
      af[i].h[1] = *(const v8bf*)&As[buf][row][16 + 8 * hi];  // K = 16..23(+8 if hi)
    }
    for (int j = 0; j < 4; j++) {
      const int col = wn + j * 16 + laneLo;
      bfrag[j].h[0] = *(const v8bf*)&Bs[buf][col][8 * hi];    // K = 0..7  (+8 if hi)
      bfrag[j].h[1] = *(const v8bf*)&Bs[buf][col][16 + 8 * hi];
    }
    for (int i = 0; i < 2; i++)
      for (int j = 0; j < 4; j++)
        acc[i][j] = __builtin_amdgcn_wmma_f32_16x16x32_bf16(
            false, af[i].v, false, bfrag[j].v, (short)0, acc[i][j], false, false);

    if (more) {
      wait_lds_fill();  // next tile resident + everyone done with current
      buf ^= 1;
    }
  }

  for (int i = 0; i < 2; i++)
    for (int j = 0; j < 4; j++)
      for (int r = 0; r < 8; r++) {
        const int m = m0 + wm + i * 16 + r + 8 * hi;
        const int n = n0 + wn + j * 16 + laneLo;
        if (F32OUT) ((float*)Cout)[(size_t)m * N + n] = acc[i][j][r];
        else        ((bf16_t*)Cout)[(size_t)m * N + n] = (bf16_t)acc[i][j][r];
      }
}

// ---------------------------------------------------------------- RMSNorm + RoPE (in place, bf16)
// grid (S, nheads), 64 threads: thread d owns pair (d, d+64) of the 128-dim head.
__global__ void rmsnorm_rope(bf16_t* __restrict__ buf, int rowStride) {
  const int s = blockIdx.x;
  const int h = blockIdx.y;
  const int d = threadIdx.x;  // 0..63
  bf16_t* p = buf + (size_t)s * rowStride + h * HEAD_DIM;
  float a = (float)p[d];
  float b = (float)p[d + 64];

  __shared__ float red[64];
  red[d] = a * a + b * b;
  __syncthreads();
  for (int off = 32; off > 0; off >>= 1) {
    if (d < off) red[d] += red[d + off];
    __syncthreads();
  }
  const float rn = rsqrtf(red[0] * (1.0f / HEAD_DIM) + 1.1920929e-07f);
  a *= rn; b *= rn;

  const float inv_freq = __expf(-(float)d * (9.210340371976184f / 64.0f));  // 10000^(-d/64)
  const float fr = (float)s * inv_freq;
  const float c = __cosf(fr), sn = __sinf(fr);
  p[d]      = (bf16_t)(a * c + b * sn);
  p[d + 64] = (bf16_t)(b * c - a * sn);
}

// ---------------------------------------------------------------- Flash attention (windowed causal)
// grid (S/64, NUM_HEADS), 128 threads = 4 waves; wave w owns query rows q0..q0+15.
__global__ __launch_bounds__(128, 1)
void attn_kernel(const bf16_t* __restrict__ qb, const bf16_t* __restrict__ kb,
                 const bf16_t* __restrict__ vb, bf16_t* __restrict__ yb) {
  __shared__ bf16_t Ks[64][136];       // [key][d], row stride 272B
  __shared__ bf16_t Vt[128][72];       // [d][key], row stride 144B
  __shared__ bf16_t Pw[4][16][72];     // per-wave P tile [qrow][key]

  const int h    = blockIdx.y;
  const int kvh  = h / (NUM_HEADS / NUM_KV_HEADS);
  const int qblk = blockIdx.x * 64;
  const int tid  = threadIdx.x;
  const int wave = tid >> 5;
  const int lane = tid & 31;
  const int laneLo = lane & 15;
  const int hi     = lane >> 4;
  const int q0 = qblk + wave * 16;

  // Q fragments (A-layout), resident for whole kernel: 4 K-chunks of 32
  V16 qf[4];
  {
    const bf16_t* qrow = qb + (size_t)(q0 + laneLo) * MODEL_DIM + h * HEAD_DIM;
    for (int c = 0; c < 4; c++) {
      qf[c].h[0] = *(const v8bf*)(qrow + c * 32 + 8 * hi);
      qf[c].h[1] = *(const v8bf*)(qrow + c * 32 + 16 + 8 * hi);
    }
  }

  float m[8], l[8];
  v8f yacc[8];
  for (int r = 0; r < 8; r++) { m[r] = -1e30f; l[r] = 0.0f; }
  for (int f = 0; f < 8; f++)
    for (int r = 0; r < 8; r++) yacc[f][r] = 0.0f;

  const float sscale = 0.08838834764831845f;  // 1/sqrt(128)

  int ktlo = qblk - WINDOW;
  if (ktlo < 0) ktlo = 0;

  for (int kt0 = ktlo; kt0 <= qblk; kt0 += 64) {
    __syncthreads();
    // K tile 64x128 (row-major), async 16B chunks
    for (int i = tid; i < 64 * 16; i += 128) {
      const int key = i >> 4, seg = i & 15;
      cp16(&Ks[key][seg * 8],
           kb + (size_t)(kt0 + key) * KV_DIM + kvh * HEAD_DIM + seg * 8);
    }
    // V tile transposed: [d][key] (element transpose -> scalar ds stores)
    {
      const bf16_t* vcol = vb + (size_t)kt0 * KV_DIM + kvh * HEAD_DIM + tid;
      for (int key = 0; key < 64; key++) Vt[tid][key] = vcol[(size_t)key * KV_DIM];
    }
    wait_lds_fill();

    // S = Q K^T : 4 key-column frags x 4 d-chunks
    v8f sacc[4];
    for (int f = 0; f < 4; f++)
      for (int r = 0; r < 8; r++) sacc[f][r] = 0.0f;
    for (int c = 0; c < 4; c++) {
      for (int f = 0; f < 4; f++) {
        V16 kfr;
        const int key = f * 16 + laneLo;
        kfr.h[0] = *(const v8bf*)&Ks[key][c * 32 + 8 * hi];
        kfr.h[1] = *(const v8bf*)&Ks[key][c * 32 + 16 + 8 * hi];
        sacc[f] = __builtin_amdgcn_wmma_f32_16x16x32_bf16(
            false, qf[c].v, false, kfr.v, (short)0, sacc[f], false, false);
      }
    }

    // mask + online softmax (C layout: row = r + 8*hi, col = lane%16)
    float rowm[8];
    for (int r = 0; r < 8; r++) rowm[r] = -1e30f;
    for (int f = 0; f < 4; f++) {
      const int key = kt0 + f * 16 + laneLo;
      for (int r = 0; r < 8; r++) {
        const int qr = q0 + r + 8 * hi;
        float sv = sacc[f][r] * sscale;
        const bool ok = (key <= qr) && (key > qr - WINDOW);
        sv = ok ? sv : -1e30f;
        sacc[f][r] = sv;
        rowm[r] = fmaxf(rowm[r], sv);
      }
    }
    for (int r = 0; r < 8; r++)
      for (int msk = 1; msk < 16; msk <<= 1)
        rowm[r] = fmaxf(rowm[r], __shfl_xor(rowm[r], msk, 32));

    float scl[8];
    for (int r = 0; r < 8; r++) {
      const float mn = fmaxf(m[r], rowm[r]);
      scl[r] = __expf(m[r] - mn);
      m[r] = mn;
      l[r] *= scl[r];
    }
    for (int f = 0; f < 8; f++)
      for (int r = 0; r < 8; r++) yacc[f][r] *= scl[r];

    float rsum[8];
    for (int r = 0; r < 8; r++) rsum[r] = 0.0f;
    for (int f = 0; f < 4; f++) {
      const int key = kt0 + f * 16 + laneLo;
      for (int r = 0; r < 8; r++) {
        const int qr = q0 + r + 8 * hi;
        const bool ok = (key <= qr) && (key > qr - WINDOW);
        const float p = ok ? __expf(sacc[f][r] - m[r]) : 0.0f;
        rsum[r] += p;
        Pw[wave][r + 8 * hi][f * 16 + laneLo] = (bf16_t)p;
      }
    }
    for (int r = 0; r < 8; r++) {
      for (int msk = 1; msk < 16; msk <<= 1) rsum[r] += __shfl_xor(rsum[r], msk, 32);
      l[r] += rsum[r];
    }
    __syncthreads();  // Pw visible (cross-lane), before P@V reads

    // Y += P V : 2 key-chunks of 32 x 8 d-column frags
    for (int c = 0; c < 2; c++) {
      V16 pa;
      pa.h[0] = *(const v8bf*)&Pw[wave][laneLo][c * 32 + 8 * hi];
      pa.h[1] = *(const v8bf*)&Pw[wave][laneLo][c * 32 + 16 + 8 * hi];
      for (int f = 0; f < 8; f++) {
        V16 vfr;
        const int d = f * 16 + laneLo;
        vfr.h[0] = *(const v8bf*)&Vt[d][c * 32 + 8 * hi];
        vfr.h[1] = *(const v8bf*)&Vt[d][c * 32 + 16 + 8 * hi];
        yacc[f] = __builtin_amdgcn_wmma_f32_16x16x32_bf16(
            false, pa.v, false, vfr.v, (short)0, yacc[f], false, false);
      }
    }
  }

  // epilogue: divide by row sums, store bf16 y (layout (s, h*128+d))
  for (int f = 0; f < 8; f++)
    for (int r = 0; r < 8; r++) {
      const int row = q0 + r + 8 * hi;
      const int col = h * HEAD_DIM + f * 16 + laneLo;
      yb[(size_t)row * MODEL_DIM + col] = (bf16_t)(yacc[f][r] / l[r]);
    }
}

// ---------------------------------------------------------------- launch
extern "C" void kernel_launch(void* const* d_in, const int* in_sizes, int n_in,
                              void* d_out, int out_size, void* d_ws, size_t ws_size,
                              hipStream_t stream) {
  (void)in_sizes; (void)n_in; (void)out_size; (void)ws_size;
  const float* x  = (const float*)d_in[0];
  const float* Wq = (const float*)d_in[1];
  const float* Wk = (const float*)d_in[2];
  const float* Wv = (const float*)d_in[3];
  const float* Wo = (const float*)d_in[4];
  float* out = (float*)d_out;

  char* ws = (char*)d_ws;
  size_t off = 0;
  auto alloc = [&](size_t bytes) -> void* {
    void* p = ws + off;
    off += (bytes + 255) & ~(size_t)255;
    return p;
  };
  bf16_t* xb   = (bf16_t*)alloc((size_t)SEQ * MODEL_DIM * 2);
  bf16_t* wqb  = (bf16_t*)alloc((size_t)MODEL_DIM * MODEL_DIM * 2);
  bf16_t* wkb  = (bf16_t*)alloc((size_t)KV_DIM * MODEL_DIM * 2);
  bf16_t* wvb  = (bf16_t*)alloc((size_t)KV_DIM * MODEL_DIM * 2);
  bf16_t* wob  = (bf16_t*)alloc((size_t)MODEL_DIM * MODEL_DIM * 2);
  bf16_t* qbuf = (bf16_t*)alloc((size_t)SEQ * MODEL_DIM * 2);
  bf16_t* kbuf = (bf16_t*)alloc((size_t)SEQ * KV_DIM * 2);
  bf16_t* vbuf = (bf16_t*)alloc((size_t)SEQ * KV_DIM * 2);
  bf16_t* ybuf = (bf16_t*)alloc((size_t)SEQ * MODEL_DIM * 2);

  auto cast = [&](const float* in, bf16_t* o, int n) {
    int blocks = (n + 256 * 8 - 1) / (256 * 8);
    if (blocks > 4096) blocks = 4096;
    cast_f32_bf16<<<blocks, 256, 0, stream>>>(in, o, n);
  };
  cast(x,  xb,  SEQ * MODEL_DIM);
  cast(Wq, wqb, MODEL_DIM * MODEL_DIM);
  cast(Wk, wkb, KV_DIM * MODEL_DIM);
  cast(Wv, wvb, KV_DIM * MODEL_DIM);
  cast(Wo, wob, MODEL_DIM * MODEL_DIM);

  gemm_bf16<false><<<dim3(MODEL_DIM / 128, SEQ / 128), 256, 0, stream>>>(
      xb, wqb, qbuf, SEQ, MODEL_DIM, MODEL_DIM);
  gemm_bf16<false><<<dim3(KV_DIM / 128, SEQ / 128), 256, 0, stream>>>(
      xb, wkb, kbuf, SEQ, KV_DIM, MODEL_DIM);
  gemm_bf16<false><<<dim3(KV_DIM / 128, SEQ / 128), 256, 0, stream>>>(
      xb, wvb, vbuf, SEQ, KV_DIM, MODEL_DIM);

  rmsnorm_rope<<<dim3(SEQ, NUM_HEADS),    64, 0, stream>>>(qbuf, MODEL_DIM);
  rmsnorm_rope<<<dim3(SEQ, NUM_KV_HEADS), 64, 0, stream>>>(kbuf, KV_DIM);

  attn_kernel<<<dim3(SEQ / 64, NUM_HEADS), 128, 0, stream>>>(qbuf, kbuf, vbuf, ybuf);

  gemm_bf16<true><<<dim3(MODEL_DIM / 128, SEQ / 128), 256, 0, stream>>>(
      ybuf, wob, out, SEQ, MODEL_DIM, MODEL_DIM);
}
